// AM2P_55113020342736
// MI455X (gfx1250) — matrix operand order repaired and checked by reading, not compiled, not run on hardware
//
#include <hip/hip_runtime.h>
#include <hip/hip_bf16.h>

#define HH 96
#define WW 96
#define HWP 9216           // H*W
#define CC 384
#define SS 4
#define BB 8
#define MM 64
#define NP 80              // padded proto count: 64 local + 1 global + 15 zero
#define BETA_ 0.3f
#define TEMP_ 0.07f
#define EPS_ 1e-6f
#define PSTRIDE 392        // bf16 per LDS proto row: 384 + 8 pad (196 dwords, %64==4 -> conflict-free b128)
#define ROWVEC 48          // 16B vectors per packed proto row (384*2/16)

typedef __attribute__((ext_vector_type(16))) __bf16 v16bf;
typedef __attribute__((ext_vector_type(8)))  __bf16 v8bf;
typedef __attribute__((ext_vector_type(8)))  float  v8f;

// ---------------- Kernel 1: local prototypes (one block per anchor) ----------------
__global__ __launch_bounds__(256)
void k_local_protos(const float* __restrict__ sf,   // [S,C,H,W]
                    const int*   __restrict__ smask,// [S,1,H,W]
                    const int*   __restrict__ apos, // [M,2] (y,x)
                    const int*   __restrict__ asamp,// [M]
                    const int*   __restrict__ arad, // [M]
                    __bf16* __restrict__ Pbf,       // [NP,C] bf16, writes rows 0..63
                    float* __restrict__ wsum_m)     // [M]
{
    __shared__ float red[256];
    const int m   = blockIdx.x;
    const int tid = threadIdx.x;
    const int y = apos[2*m], x = apos[2*m+1];
    const int s = asamp[m];
    const int ri = arad[m];
    const int r = (ri == 0) ? 4 : ((ri == 1) ? 8 : 16);
    const int ylo = max(0, y - r), yhi = min(HH - 1, y + r);
    const int xlo = max(0, x - r), xhi = min(WW - 1, x + r);

    const int* mrow = smask + (size_t)s * HWP;
    const int c0 = tid, c1 = tid + 256;
    const float* f0p = sf + ((size_t)s * CC + c0) * HWP;
    const float* f1p = sf + ((size_t)s * CC + (c1 < CC ? c1 : 0)) * HWP;

    float f0 = 0.f, f1 = 0.f, msum = 0.f;
    for (int yy = ylo; yy <= yhi; ++yy) {
        const int rowp = yy * WW;
        for (int xx = xlo; xx <= xhi; ++xx) {
            const int p = rowp + xx;
            const int mk = mrow[p];
            msum += (float)mk;
            if (mk) {
                f0 += f0p[p];
                if (c1 < CC) f1 += f1p[p];
            }
        }
    }
    const float inv_m = 1.0f / (msum + EPS_);
    const float mean0 = f0 * inv_m;
    const float mean1 = f1 * inv_m;

    float sq = mean0 * mean0 + ((c1 < CC) ? mean1 * mean1 : 0.f);
    red[tid] = sq; __syncthreads();
    for (int off = 128; off > 0; off >>= 1) {
        if (tid < off) red[tid] += red[tid + off];
        __syncthreads();
    }
    const float inv = 1.0f / fmaxf(sqrtf(red[0]), 1e-12f);
    Pbf[(size_t)m * CC + c0] = (__bf16)(mean0 * inv);
    if (c1 < CC) Pbf[(size_t)m * CC + c1] = (__bf16)(mean1 * inv);
    if (tid == 0) wsum_m[m] = msum;
}

// ---------------- Kernel 2: masked global sum per channel ----------------
__global__ __launch_bounds__(256)
void k_global_sum(const float* __restrict__ sf, const int* __restrict__ smask,
                  float* __restrict__ gsum)
{
    __shared__ float red[256];
    const int c = blockIdx.x;
    const int tid = threadIdx.x;
    float acc = 0.f;
    for (int s = 0; s < SS; ++s) {
        const float* fp = sf + ((size_t)s * CC + c) * HWP;
        const int*   mp = smask + (size_t)s * HWP;
        for (int p = tid; p < HWP; p += 256)
            acc += fp[p] * (float)mp[p];
    }
    red[tid] = acc; __syncthreads();
    for (int off = 128; off > 0; off >>= 1) {
        if (tid < off) red[tid] += red[tid + off];
        __syncthreads();
    }
    if (tid == 0) gsum[c] = red[0];
}

// ---------------- Kernel 3: normalize global proto, pad, fold weights ----------------
__global__ __launch_bounds__(256)
void k_finalize(const float* __restrict__ gsum, const float* __restrict__ wsum_m,
                __bf16* __restrict__ Pbf, float* __restrict__ wvec)
{
    __shared__ float red[256];
    const int tid = threadIdx.x;

    float sq = 0.f;
    for (int c = tid; c < CC; c += 256) sq += gsum[c] * gsum[c];
    red[tid] = sq; __syncthreads();
    for (int off = 128; off > 0; off >>= 1) {
        if (tid < off) red[tid] += red[tid + off];
        __syncthreads();
    }
    const float ginv = 1.0f / fmaxf(sqrtf(red[0]), 1e-12f);
    for (int c = tid; c < CC; c += 256) {
        Pbf[(size_t)MM * CC + c] = (__bf16)(gsum[c] * ginv);   // proto 64 = global
        for (int n = MM + 1; n < NP; ++n)
            Pbf[(size_t)n * CC + c] = (__bf16)0.0f;            // zero pad rows
    }
    __syncthreads();

    float wacc = (tid < MM) ? wsum_m[tid] : 0.f;
    red[tid] = wacc; __syncthreads();
    for (int off = 128; off > 0; off >>= 1) {
        if (tid < off) red[tid] += red[tid + off];
        __syncthreads();
    }
    const float tot    = red[0] + EPS_;       // sum(msum)+eps
    const float s1     = red[0] / tot;        // sum of first-normalized weights
    const float denom2 = s1 + EPS_;           // second normalization in reference()
    if (tid < NP) {
        float wv;
        if (tid < MM)       wv = (1.0f - BETA_) * ((wsum_m[tid] / tot) / denom2);
        else if (tid == MM) wv = BETA_;
        else                wv = 0.f;
        wvec[tid] = wv;
    }
}

// ---------------- Kernel 4: fused bf16-WMMA similarity + logits ----------------
__global__ __launch_bounds__(256)
void k_fused(const float*  __restrict__ q,    // [B,C,H,W]
             const __bf16* __restrict__ Pbf,  // [NP,C] bf16 packed (normalized protos)
             const float*  __restrict__ wvec, // [NP]
             float* __restrict__ out)         // [B,2,H,W]
{
    __shared__ __bf16 Plds[NP * PSTRIDE];
    __shared__ float  wlds[NP];
    __shared__ float  strans[8 * 16];

    const int tid = threadIdx.x;

    // ---- async-stage packed bf16 protos into padded LDS layout (16B granules) ----
    {
        const unsigned ldsbase = (unsigned)(size_t)&Plds[0];
        const unsigned long long gbase = (unsigned long long)(size_t)Pbf;
        for (int v = tid; v < NP * ROWVEC; v += 256) {           // 3840 vectors, 15/thread
            const int n   = v / ROWVEC;
            const int rem = v - n * ROWVEC;
            const unsigned ldsoff = ldsbase + (unsigned)(n * (PSTRIDE * 2) + rem * 16);
            const unsigned goff   = (unsigned)v * 16u;
            asm volatile("global_load_async_to_lds_b128 %0, %1, %2"
                         :: "v"(ldsoff), "v"(goff), "s"(gbase)
                         : "memory");
        }
    }
    if (tid < NP) wlds[tid] = wvec[tid];
    asm volatile("s_wait_asynccnt 0x0" ::: "memory");
    __syncthreads();

    const int wave = tid >> 5;
    const int lane = tid & 31;
    const int hf   = lane >> 4;     // half: 0 or 1
    const int nl   = lane & 15;

    const int t  = blockIdx.x * 8 + wave;   // 0..4607
    const int bi = t / 576;                 // image
    const int p0 = (t - bi * 576) * 16;     // first pixel of tile

    const float* qbase = q + (size_t)bi * CC * HWP + p0 + nl;

    v8f acc0 = {}, acc1 = {}, acc2 = {}, acc3 = {}, acc4 = {};
    float qqa = 0.f, qqb = 0.f, qqc = 0.f, qqd = 0.f;

    for (int k0 = 0; k0 < CC; k0 += 32) {
        // ---- A fragment: 16 pixels x 32 channels, ISA 16-bit A layout ----
        const float* ap = qbase + (size_t)(k0 + hf * 8) * HWP;
        v16bf a;
        #pragma unroll
        for (int j = 0; j < 16; ++j) {
            const int cj = (j < 8) ? j : (j + 8);   // K = k0 + hf*8 + cj
            const float v = ap[(size_t)cj * HWP];
            if      ((j & 3) == 0) qqa += v * v;
            else if ((j & 3) == 1) qqb += v * v;
            else if ((j & 3) == 2) qqc += v * v;
            else                   qqd += v * v;
            a[j] = (__bf16)v;
        }
        // ---- B fragments from LDS (K x N tiles, N = tile*16 + nl) ----
        const __bf16* bb = &Plds[(size_t)nl * PSTRIDE + k0 + hf * 8];
        #define LOADB(T, OFS) \
            v8bf lo##T = *(const v8bf*)(bb + (OFS) * 16 * PSTRIDE); \
            v8bf hi##T = *(const v8bf*)(bb + (OFS) * 16 * PSTRIDE + 16); \
            v16bf bf##T = __builtin_shufflevector(lo##T, hi##T, 0,1,2,3,4,5,6,7,8,9,10,11,12,13,14,15);
        LOADB(0, 0) LOADB(1, 1) LOADB(2, 2) LOADB(3, 3) LOADB(4, 4)
        #undef LOADB

        acc0 = __builtin_amdgcn_wmma_f32_16x16x32_bf16(false, a, false, bf0, (short)0, acc0, false, false);
        acc1 = __builtin_amdgcn_wmma_f32_16x16x32_bf16(false, a, false, bf1, (short)0, acc1, false, false);
        acc2 = __builtin_amdgcn_wmma_f32_16x16x32_bf16(false, a, false, bf2, (short)0, acc2, false, false);
        acc3 = __builtin_amdgcn_wmma_f32_16x16x32_bf16(false, a, false, bf3, (short)0, acc3, false, false);
        acc4 = __builtin_amdgcn_wmma_f32_16x16x32_bf16(false, a, false, bf4, (short)0, acc4, false, false);
    }
    float qq = (qqa + qqb) + (qqc + qqd);

    // weighted combine over N (this lane's N column in each tile)
    const float w0 = wlds[nl], w1 = wlds[16 + nl], w2 = wlds[32 + nl],
                w3 = wlds[48 + nl], w4 = wlds[64 + nl];
    float part[8];
    #pragma unroll
    for (int i = 0; i < 8; ++i)
        part[i] = acc0[i] * w0 + acc1[i] * w1 + acc2[i] * w2 + acc3[i] * w3 + acc4[i] * w4;

    // reduce across the 16 lanes of each half (C/D layout: N = lane%16)
    #pragma unroll
    for (int i = 0; i < 8; ++i) {
        float v = part[i];
        v += __shfl_xor(v, 1);
        v += __shfl_xor(v, 2);
        v += __shfl_xor(v, 4);
        v += __shfl_xor(v, 8);
        part[i] = v;    // s for pixel M = i + 8*hf (same across the half)
    }
    // lane j and j+16 jointly cover all channels of pixel M=j
    qq += __shfl_xor(qq, 16);

    // transpose s[M] via LDS so lane nl gets pixel p0+nl
    float* st = &strans[wave * 16];
    if (nl < 8) st[hf * 8 + nl] = part[nl];
    __syncthreads();

    if (lane < 16) {
        const float sM   = st[nl];
        const float inrm = 1.0f / fmaxf(sqrtf(qq), 1e-12f);
        const float sfg  = sM * inrm * (1.0f / TEMP_);
        const int p = p0 + nl;
        out[((size_t)bi * 2 + 0) * HWP + p] = -sfg;
        out[((size_t)bi * 2 + 1) * HWP + p] =  sfg;
    }
}

extern "C" void kernel_launch(void* const* d_in, const int* in_sizes, int n_in,
                              void* d_out, int out_size, void* d_ws, size_t ws_size,
                              hipStream_t stream) {
    const float* sf    = (const float*)d_in[0];  // support_feats [4,384,96,96]
    const int*   smask = (const int*)  d_in[1];  // support_masks [4,1,96,96]
    const float* qf    = (const float*)d_in[2];  // query_feats   [8,384,96,96]
    const int*   apos  = (const int*)  d_in[3];  // anchor_pos    [64,2]
    const int*   asamp = (const int*)  d_in[4];  // anchor_sample [64]
    const int*   arad  = (const int*)  d_in[5];  // anchor_radius [64]
    float* out = (float*)d_out;                  // [8,2,96,96]

    __bf16* wsPbf = (__bf16*)d_ws;                          // [NP*CC] bf16 packed
    float*  wsW   = (float*)((char*)d_ws + NP * CC * 2);    // [NP]    folded weights
    float*  wsG   = wsW + NP;                               // [CC]    masked global sums
    float*  wsM   = wsG + CC;                               // [MM]    per-anchor mask sums

    k_local_protos<<<MM, 256, 0, stream>>>(sf, smask, apos, asamp, arad, wsPbf, wsM);
    k_global_sum  <<<CC, 256, 0, stream>>>(sf, smask, wsG);
    k_finalize    <<<1,  256, 0, stream>>>(wsG, wsM, wsPbf, wsW);
    k_fused       <<<576, 256, 0, stream>>>(qf, wsPbf, wsW, out);
}
